// TriangularSelfAttentionBlock_58059367908129
// MI455X (gfx1250) — compile-verified, matrix operand order checked by reading, not run on hardware
//
#include <hip/hip_runtime.h>
#include <hip/hip_bf16.h>
#include <cstdint>

// ===================================================================
// TriangularSelfAttentionBlock for MI455X (gfx1250, wave32, WMMA bf16)
//   - all GEMMs (linears + triangle einsums) via v_wmma_f32_16x16x32_bf16
//   - fused online-softmax attention (head dim 32 == one WMMA K step),
//     cooperative K/V staging in LDS, two query tiles per wave
//   - fp32 activations in HBM/L2 (pair state fits the 192MB L2),
//     bf16 packed pairs inside the MMA pipeline (b128 loads/stores)
// ===================================================================

typedef __attribute__((ext_vector_type(16))) __bf16 v16bf;
typedef __attribute__((ext_vector_type(8)))  float  v8f;

#define ACT_NONE 0
#define ACT_SIG  1
#define ACT_RELU 2
#define ATT_INF  1e9f

__device__ __forceinline__ unsigned short f32_bf16(float f) {
  union { float f; unsigned u; } v; v.f = f;
  return (unsigned short)((v.u + 0x7FFFu + ((v.u >> 16) & 1u)) >> 16);  // RNE
}
// pack two fp32 -> dword of two bf16 (lo = a, hi = b)
__device__ __forceinline__ unsigned pack_bf16(float a, float b) {
  union { float f; unsigned u; } x, y; x.f = a; y.f = b;
  unsigned ra = x.u + 0x7FFFu + ((x.u >> 16) & 1u);
  unsigned rb = y.u + 0x7FFFu + ((y.u >> 16) & 1u);
  return (ra >> 16) | (rb & 0xFFFF0000u);
}
__device__ __forceinline__ float sigmoidf_(float x) { return 1.f / (1.f + __expf(-x)); }
// K-index of fragment element pair 2v within a lane half (A and B share this striping)
__device__ __forceinline__ int kfrag(int v, int hi) {
  return (v < 4 ? 2 * v : 16 + 2 * (v - 4)) + 8 * hi;
}
union Frag { unsigned u[8]; v16bf v; };

// ---------------- LayerNorm: one wave per row ----------------------
__global__ __launch_bounds__(256) void ln_kernel(
    const float* __restrict__ x, const float* __restrict__ g, const float* __restrict__ b,
    float* __restrict__ out, int M, int D) {
  int wave = threadIdx.x >> 5, lane = threadIdx.x & 31;
  int row = blockIdx.x * 8 + wave;
  if (row >= M) return;
  const float* xr = x + (size_t)row * D;
  float s = 0.f, ss = 0.f;
  for (int i = lane; i < D; i += 32) { float v = xr[i]; s += v; ss += v * v; }
  for (int o = 16; o; o >>= 1) { s += __shfl_xor(s, o, 32); ss += __shfl_xor(ss, o, 32); }
  float mean = s / D;
  float inv = rsqrtf(ss / D - mean * mean + 1e-5f);
  float* orow = out + (size_t)row * D;
  for (int i = lane; i < D; i += 32) orow[i] = (xr[i] - mean) * inv * g[i] + b[i];
}

// ---------------- bf16 WMMA GEMM:  C[M,N] = act(A[M,K] @ W[N,K]^T + bias) (+res)
// 64x64 tile, K-stage 64, 8 waves (2x4), each wave 32x16 (two 16x16 accumulators)
// Block-uniform full-tile fast path -> unconditional global_load_b128.
__global__ __launch_bounds__(256) void gemm_kernel(
    const float* __restrict__ A, const float* __restrict__ W,
    const float* __restrict__ bias, const float* __restrict__ res,
    float* __restrict__ C, int M, int N, int K,
    long aBatch, long wBatch, long cBatch, int act) {
  __shared__ unsigned As[64][36];   // 64 rows x 64 k as bf16 pairs (32 dwords) + 4 pad
  __shared__ unsigned Bs[64][36];
  int bz = blockIdx.z;
  A += (size_t)bz * aBatch; W += (size_t)bz * wBatch; C += (size_t)bz * cBatch;
  if (res) res += (size_t)bz * cBatch;
  int m0 = blockIdx.y * 64, n0 = blockIdx.x * 64;
  int tid = threadIdx.x;
  int wave = tid >> 5, lane = tid & 31, l15 = lane & 15, hi = lane >> 4;
  int wm = wave >> 2, wn = wave & 3;
  v8f acc0 = {}, acc1 = {};
  int ldr = tid >> 2;            // 0..63: tile row this thread loads
  int ldk = (tid & 3) * 16;      // 0,16,32,48: start of its 16-wide k chunk
  int gm = m0 + ldr, gn = n0 + ldr;
  bool fullMN = (m0 + 64 <= M) && (n0 + 64 <= N);   // block-uniform
  for (int k0 = 0; k0 < K; k0 += 64) {
    // speculative prefetch of the next stage (invalid addresses silently dropped)
    __builtin_prefetch(A + (size_t)gm * K + k0 + 64 + ldk, 0, 1);
    __builtin_prefetch(W + (size_t)gn * K + k0 + 64 + ldk, 0, 1);
    float4 a4[4], w4[4];
    if (fullMN && (k0 + 64 <= K)) {                 // fast path: wide unconditional loads
      const float4* ap = (const float4*)(A + (size_t)gm * K + k0 + ldk);
      const float4* wp = (const float4*)(W + (size_t)gn * K + k0 + ldk);
#pragma unroll
      for (int q = 0; q < 4; ++q) { a4[q] = ap[q]; w4[q] = wp[q]; }
    } else {                                        // edge blocks only
      bool mok = gm < M, nok = gn < N;
#pragma unroll
      for (int q = 0; q < 4; ++q) {
        int kg = k0 + ldk + q * 4;
        float4 z4 = {0.f, 0.f, 0.f, 0.f};
        bool kok = kg < K;
        a4[q] = (mok && kok) ? *(const float4*)(A + (size_t)gm * K + kg) : z4;
        w4[q] = (nok && kok) ? *(const float4*)(W + (size_t)gn * K + kg) : z4;
      }
    }
#pragma unroll
    for (int q = 0; q < 4; ++q) {
      int d = (ldk >> 1) + q * 2;
      As[ldr][d] = pack_bf16(a4[q].x, a4[q].y); As[ldr][d + 1] = pack_bf16(a4[q].z, a4[q].w);
      Bs[ldr][d] = pack_bf16(w4[q].x, w4[q].y); Bs[ldr][d + 1] = pack_bf16(w4[q].z, w4[q].w);
    }
    __syncthreads();
#pragma unroll
    for (int kc = 0; kc < 2; ++kc) {          // two 32-wide K chunks per stage
      if (kc == 1 && k0 + 32 >= K) break;
      int kb = kc * 16;                        // dword base within LDS row
      Frag ub;
      int br = wn * 16 + l15;
#pragma unroll
      for (int v = 0; v < 4; ++v) {
        ub.u[v]     = Bs[br][kb + v + 4 * hi];
        ub.u[v + 4] = Bs[br][kb + 8 + v + 4 * hi];
      }
#pragma unroll
      for (int half = 0; half < 2; ++half) {
        Frag ua;
        int ar = wm * 32 + half * 16 + l15;
#pragma unroll
        for (int v = 0; v < 4; ++v) {
          ua.u[v]     = As[ar][kb + v + 4 * hi];
          ua.u[v + 4] = As[ar][kb + 8 + v + 4 * hi];
        }
        if (half == 0)
          acc0 = __builtin_amdgcn_wmma_f32_16x16x32_bf16(false, ua.v, false, ub.v, (short)0, acc0, false, false);
        else
          acc1 = __builtin_amdgcn_wmma_f32_16x16x32_bf16(false, ua.v, false, ub.v, (short)0, acc1, false, false);
      }
    }
    __syncthreads();
  }
  int n = n0 + wn * 16 + l15;
  if (n < N) {                                  // one predicate per lane
    float bv = bias ? bias[n] : 0.f;
    bool fullM = (m0 + 64 <= M);
#pragma unroll
    for (int half = 0; half < 2; ++half) {
      v8f acc = acc0; if (half) acc = acc1;
#pragma unroll
      for (int v = 0; v < 8; ++v) {
        int m = m0 + wm * 32 + half * 16 + v + 8 * hi;
        if (fullM || m < M) {
          float x = acc[v] + bv;
          if (act == ACT_SIG) x = sigmoidf_(x);
          else if (act == ACT_RELU) x = fmaxf(x, 0.f);
          size_t off = (size_t)m * N + n;
          if (res) x += res[off];
          C[off] = x;
        }
      }
    }
  }
}

// ---------------- fused attention, head dim 32, seqlen 256 ---------
// batch = outer*heads + h.  O[q,c] = softmax_k(Q·K^T*scale + bias + maskbias) V
// Cooperative K/V staging per 32-key block; each wave owns TWO query tiles.
__global__ __launch_bounds__(256) void attn_kernel(
    const float* __restrict__ Qp, const float* __restrict__ Kp, const float* __restrict__ Vp,
    const float* __restrict__ Bp, const unsigned char* __restrict__ maskp,
    float* __restrict__ Op, int heads,
    long q_outer, long q_head, long q_row,
    long b_outer, long b_head, long b_row, long b_key,
    long o_outer, long o_head, long o_row,
    float scale, int Lk, int use_outer_mask) {
  __shared__ unsigned Ks[32][20];             // [key][ch bf16-pairs] + pad
  __shared__ unsigned short Vt[32][40];       // [ch][key] bf16, key pairs dword-aligned
  __shared__ unsigned short Plds[8][16][34];  // per-wave P staging
  int bi = blockIdx.x;
  int h = bi % heads, outer = bi / heads;
  const float* Q = Qp + (size_t)outer * q_outer + (size_t)h * q_head;
  const float* Km = Kp + (size_t)outer * q_outer + (size_t)h * q_head;
  const float* V = Vp + (size_t)outer * q_outer + (size_t)h * q_head;
  const float* B = Bp + (size_t)outer * b_outer + (size_t)h * b_head;
  float* O = Op + (size_t)outer * o_outer + (size_t)h * o_head;
  int tid = threadIdx.x;
  int wave = tid >> 5, lane = tid & 31, l15 = lane & 15, hi = lane >> 4;
  float fm_outer = (maskp && use_outer_mask) ? (maskp[outer] ? 1.f : 0.f) : 1.f;

  // two query tiles per wave: qbase = (wave + 8t)*16
  v16bf qf[2];
  float rowmax[2][8], rowsum[2][8];
  v8f out00 = {}, out01 = {}, out10 = {}, out11 = {};
#pragma unroll
  for (int t = 0; t < 2; ++t) {
    int qbase = (wave + 8 * t) * 16;
    const float* qr = Q + (size_t)(qbase + l15) * q_row;
    Frag uq;
#pragma unroll
    for (int v = 0; v < 8; ++v) {
      int c = kfrag(v, hi);
      float2 q2 = *(const float2*)(qr + c);
      uq.u[v] = pack_bf16(q2.x * scale, q2.y * scale);
    }
    qf[t] = uq.v;
#pragma unroll
    for (int v = 0; v < 8; ++v) { rowmax[t][v] = -3.0e38f; rowsum[t][v] = 0.f; }
  }

  int ldr = tid >> 3;        // 0..31: key row this thread stages
  int ldc = (tid & 7) * 4;   // channel chunk 0,4,...,28

  for (int kb = 0; kb < Lk; kb += 32) {
    __syncthreads();
    {
      float4 k4 = *(const float4*)(Km + (size_t)(kb + ldr) * q_row + ldc);
      Ks[ldr][(ldc >> 1)]     = pack_bf16(k4.x, k4.y);
      Ks[ldr][(ldc >> 1) + 1] = pack_bf16(k4.z, k4.w);
      float4 v4 = *(const float4*)(V + (size_t)(kb + ldr) * q_row + ldc);
      Vt[ldc + 0][ldr] = f32_bf16(v4.x);
      Vt[ldc + 1][ldr] = f32_bf16(v4.y);
      Vt[ldc + 2][ldr] = f32_bf16(v4.z);
      Vt[ldc + 3][ldr] = f32_bf16(v4.w);
    }
    __syncthreads();
    // build K fragments (scores) and V fragments (PV) once, shared by both tiles
    v16bf kf[2], vf[2];
#pragma unroll
    for (int half = 0; half < 2; ++half) {
      Frag uk, uv;
      int krow = half * 16 + l15;
#pragma unroll
      for (int v = 0; v < 4; ++v) {
        uk.u[v]     = Ks[krow][v + 4 * hi];
        uk.u[v + 4] = Ks[krow][8 + v + 4 * hi];
      }
      kf[half] = uk.v;
      const unsigned* vrow = (const unsigned*)&Vt[l15 + 16 * half][0];
#pragma unroll
      for (int v = 0; v < 4; ++v) {
        uv.u[v]     = vrow[v + 4 * hi];
        uv.u[v + 4] = vrow[8 + v + 4 * hi];
      }
      vf[half] = uv.v;
    }
    int key0 = kb + l15, key1 = kb + 16 + l15;
    float fmk0 = maskp ? (maskp[key0] ? 1.f : 0.f) : 1.f;
    float fmk1 = maskp ? (maskp[key1] ? 1.f : 0.f) : 1.f;
    float mb0 = use_outer_mask ? ATT_INF * (fm_outer * fmk0 - 1.f) : ATT_INF * (fmk0 - 1.f);
    float mb1 = use_outer_mask ? ATT_INF * (fm_outer * fmk1 - 1.f) : ATT_INF * (fmk1 - 1.f);

#pragma unroll
    for (int t = 0; t < 2; ++t) {
      int qbase = (wave + 8 * t) * 16;
      v8f S0 = {}, S1 = {};
      S0 = __builtin_amdgcn_wmma_f32_16x16x32_bf16(false, qf[t], false, kf[0], (short)0, S0, false, false);
      S1 = __builtin_amdgcn_wmma_f32_16x16x32_bf16(false, qf[t], false, kf[1], (short)0, S1, false, false);
#pragma unroll
      for (int v = 0; v < 8; ++v) {
        size_t q = qbase + v + 8 * hi;
        S0[v] += B[q * b_row + (size_t)key0 * b_key] + mb0;
        S1[v] += B[q * b_row + (size_t)key1 * b_key] + mb1;
      }
      // ---- online softmax (row m = v + 8*hi, cols across 16-lane group) ----
#pragma unroll
      for (int v = 0; v < 8; ++v) {
        float m = fmaxf(S0[v], S1[v]);
#pragma unroll
        for (int o = 8; o; o >>= 1) m = fmaxf(m, __shfl_xor(m, o, 16));
        float mnew = fmaxf(rowmax[t][v], m);
        float corr = __expf(rowmax[t][v] - mnew);
        rowmax[t][v] = mnew;
        float e0 = __expf(S0[v] - mnew), e1 = __expf(S1[v] - mnew);
        S0[v] = e0; S1[v] = e1;
        float ls = e0 + e1;
#pragma unroll
        for (int o = 8; o; o >>= 1) ls += __shfl_xor(ls, o, 16);
        rowsum[t][v] = rowsum[t][v] * corr + ls;
        if (t == 0) { out00[v] *= corr; out01[v] *= corr; }
        else        { out10[v] *= corr; out11[v] *= corr; }
      }
      // stage P (C layout) -> LDS -> reload as A fragment (row-per-lane)
#pragma unroll
      for (int v = 0; v < 8; ++v) {
        int q = v + 8 * hi;
        Plds[wave][q][l15]      = f32_bf16(S0[v]);
        Plds[wave][q][l15 + 16] = f32_bf16(S1[v]);
      }
      v16bf pf;
#pragma unroll
      for (int v = 0; v < 8; ++v) {
        int k = kfrag(v, hi);
        union { unsigned short u; __bf16 b; } e0, e1;
        e0.u = Plds[wave][l15][k]; e1.u = Plds[wave][l15][k + 1];
        pf[2 * v] = e0.b; pf[2 * v + 1] = e1.b;
      }
      if (t == 0) {
        out00 = __builtin_amdgcn_wmma_f32_16x16x32_bf16(false, pf, false, vf[0], (short)0, out00, false, false);
        out01 = __builtin_amdgcn_wmma_f32_16x16x32_bf16(false, pf, false, vf[1], (short)0, out01, false, false);
      } else {
        out10 = __builtin_amdgcn_wmma_f32_16x16x32_bf16(false, pf, false, vf[0], (short)0, out10, false, false);
        out11 = __builtin_amdgcn_wmma_f32_16x16x32_bf16(false, pf, false, vf[1], (short)0, out11, false, false);
      }
    }
  }
#pragma unroll
  for (int t = 0; t < 2; ++t) {
    int qbase = (wave + 8 * t) * 16;
    v8f o0 = t ? out10 : out00, o1 = t ? out11 : out01;
#pragma unroll
    for (int v = 0; v < 8; ++v) {
      int q = qbase + v + 8 * hi;
      float inv = 1.f / rowsum[t][v];
      O[(size_t)q * o_row + l15]      = o0[v] * inv;
      O[(size_t)q * o_row + l15 + 16] = o1[v] * inv;
    }
  }
}

// ---------------- elementwise helpers ------------------------------
__global__ void ew_mul_kernel(const float* a, const float* b, float* c, long n) {
  for (long i = (long)blockIdx.x * blockDim.x + threadIdx.x; i < n; i += (long)gridDim.x * blockDim.x)
    c[i] = a[i] * b[i];
}
__global__ void ew_mul_add_kernel(const float* r, const float* a, const float* b, float* c, long n) {
  for (long i = (long)blockIdx.x * blockDim.x + threadIdx.x; i < n; i += (long)gridDim.x * blockDim.x)
    c[i] = r[i] + a[i] * b[i];
}
// out[i,k,c] = fm[i]*fm[k]*g[i,k,c]*p[i,k,c]
__global__ void maskmul_kernel(const float* g, const float* p, const unsigned char* mask,
                               float* out, long n, int L, int D) {
  for (long i = (long)blockIdx.x * blockDim.x + threadIdx.x; i < n; i += (long)gridDim.x * blockDim.x) {
    long t = i / D; int kk = (int)(t % L); int ii = (int)(t / L);
    float fm = mask ? ((mask[ii] ? 1.f : 0.f) * (mask[kk] ? 1.f : 0.f)) : 1.f;
    out[i] = fm * g[i] * p[i];
  }
}
// dst[i0,i1,i2] = src[i0*s0 + i1*s1 + i2*s2]  (generic pack/unpack/transpose)
__global__ void gather_kernel(float* __restrict__ dst, const float* __restrict__ src,
                              int D0, int D1, int D2, long s0, long s1, long s2) {
  long n = (long)D0 * D1 * D2;
  for (long i = (long)blockIdx.x * blockDim.x + threadIdx.x; i < n; i += (long)gridDim.x * blockDim.x) {
    int i2 = (int)(i % D2); long t = i / D2; int i1 = (int)(t % D1); int i0 = (int)(t / D1);
    dst[i] = src[(long)i0 * s0 + (long)i1 * s1 + (long)i2 * s2];
  }
}
// cat[i,j,c<64] = q[j,c]*k[i,c];  cat[i,j,c>=64] = q[j,c-64]-k[i,c-64]; q=qk[:, :64], k=qk[:, 64:]
__global__ void s2p_cat_kernel(const float* qk, float* cat, int L, int D) {
  long n = (long)L * L * D;
  for (long i = (long)blockIdx.x * blockDim.x + threadIdx.x; i < n; i += (long)gridDim.x * blockDim.x) {
    int c = (int)(i % D); long t = i / D; int j = (int)(t % L); int ii = (int)(t / L);
    int cc = c & 63;
    float q = qk[(size_t)j * D + cc];
    float k = qk[(size_t)ii * D + 64 + cc];
    cat[i] = (c < 64) ? q * k : q - k;
  }
}
// z[j,i,c] += t[i,j,c]
__global__ void tadd_kernel(float* z, const float* t, int L, int D) {
  long n = (long)L * L * D;
  for (long i = (long)blockIdx.x * blockDim.x + threadIdx.x; i < n; i += (long)gridDim.x * blockDim.x) {
    int c = (int)(i % D); long tt = i / D; int j = (int)(tt % L); int ii = (int)(tt / L);
    z[((size_t)j * L + ii) * D + c] += t[i];
  }
}

// ===================================================================
// Host orchestration
// Input flattening assumed = setup_inputs() dict insertion order,
// recursing into nested dicts depth-first (83 leaves total).
// ===================================================================
extern "C" void kernel_launch(void* const* d_in, const int* in_sizes, int n_in,
                              void* d_out, int out_size, void* d_ws, size_t ws_size,
                              hipStream_t stream) {
  (void)in_sizes; (void)n_in; (void)out_size; (void)ws_size;
  const int L = 256, DS = 1024, DP = 128;
  const long LL = (long)L * L;              // 65536
  const float ISQ32 = 0.17677669529663689f; // 1/sqrt(32)

  const float* s_in = (const float*)d_in[0];
  const float* z_in = (const float*)d_in[1];
  const unsigned char* mask = (const unsigned char*)d_in[2];
  auto P = [&](int i) { return (const float*)d_in[i]; };
  enum {
    LN1_G = 3, LN1_B, P2S_LNG, P2S_LNB, P2S_W,
    AT_PROJ, AT_OW, AT_OB, AT_GW, AT_GB,
    S2P_LNG, S2P_LNB, S2P_PW, S2P_PB, S2P_OW, S2P_OB,
    TMO = 19, TMI = 35, TAS = 51, TAE = 61, MLPS = 71, MLPP = 77
  };
  // tri_mul offsets: 0 ln_in.g 1 ln_in.b 2 ln_out.g 3 ln_out.b 4 ap_w 5 ap_b
  //                  6 ag_w 7 ag_b 8 bp_w 9 bp_b 10 bg_w 11 bg_b 12 g_w 13 g_b 14 z_w 15 z_b
  // tri_att offsets: 0 ln.g 1 ln.b 2 bias_w 3 q_w 4 k_w 5 v_w 6 g_w 7 g_b 8 o_w 9 o_b
  // mlp offsets:     0 ln.g 1 ln.b 2 w1 3 b1 4 w2 5 b2

  float* base = (float*)d_ws;
  const size_t REG = (size_t)LL * DP;       // 32MiB fp32 regions
  float* zb = base;
  float* R1 = base + REG * 1;  float* R2 = base + REG * 2;
  float* R3 = base + REG * 3;  float* R4 = base + REG * 4;
  float* R5 = base + REG * 5;  float* R6 = base + REG * 6;
  float* R7 = base + REG * 7;
  float* BIG = base + REG * 8;              // [65536,512] mlp_p hidden
  float* SEQ = base + REG * 12;
  float* y    = SEQ;
  float* tq   = y   + (size_t)L * DS;
  float* gb   = tq  + (size_t)L * 3 * DS;
  float* yat  = gb  + (size_t)L * DS;
  float* gy   = yat + (size_t)L * DS;
  float* sn   = gy  + (size_t)L * DS;
  float* qk   = sn  + (size_t)L * DS;
  float* hsq  = qk  + (size_t)L * DP;
  float* sbuf = hsq + (size_t)L * 4 * DS;

  float* sout = (float*)d_out;
  float* zout = sout + (size_t)L * DS;

  auto ln = [&](const float* x, int gi, int bi, float* o, int M, int D) {
    ln_kernel<<<dim3((M + 7) / 8), dim3(256), 0, stream>>>(x, P(gi), P(bi), o, M, D);
  };
  auto gemm = [&](const float* A, const float* W, const float* bias, const float* res,
                  float* C, int M, int N, int K, int act,
                  int batch, long aB, long wB, long cB) {
    gemm_kernel<<<dim3((N + 63) / 64, (M + 63) / 64, batch), dim3(256), 0, stream>>>(
        A, W, bias, res, C, M, N, K, aB, wB, cB, act);
  };
  auto lin = [&](const float* A, int wi, int bi, const float* res, float* C,
                 int M, int N, int K, int act) {
    gemm(A, P(wi), bi >= 0 ? P(bi) : nullptr, res, C, M, N, K, act, 1, 0, 0, 0);
  };
  const dim3 EWG(4096), EWB(256);

  // ================= sequence path =================
  ln(z_in, P2S_LNG, P2S_LNB, R1, (int)LL, DP);
  lin(R1, P2S_W, -1, nullptr, R7, (int)LL, 32, DP, ACT_NONE);       // bias[l,m,h]
  ln(s_in, LN1_G, LN1_B, y, L, DS);
  lin(y, AT_PROJ, -1, nullptr, tq, L, 3 * DS, DS, ACT_NONE);        // [l,h,(q|k|v)32]
  lin(y, AT_GW, AT_GB, nullptr, gb, L, DS, DS, ACT_SIG);
  attn_kernel<<<dim3(32), dim3(256), 0, stream>>>(
      tq + 0, tq + 32, tq + 64, R7, mask, yat, /*heads=*/32,
      0, 96, 3072,
      0, 1, (long)L * 32, 32,
      0, 32, DS, ISQ32, L, 0);
  ew_mul_kernel<<<EWG, EWB, 0, stream>>>(gb, yat, gy, (long)L * DS);
  lin(gy, AT_OW, AT_OB, s_in, sbuf, L, DS, DS, ACT_NONE);           // s = s + attn
  ln(sbuf, MLPS + 0, MLPS + 1, y, L, DS);
  lin(y, MLPS + 2, MLPS + 3, nullptr, hsq, L, 4 * DS, DS, ACT_RELU);
  lin(hsq, MLPS + 4, MLPS + 5, sbuf, sout, L, DS, 4 * DS, ACT_NONE);

  // ================= sequence -> pair =================
  ln(sout, S2P_LNG, S2P_LNB, sn, L, DS);
  lin(sn, S2P_PW, S2P_PB, nullptr, qk, L, DP, DS, ACT_NONE);
  s2p_cat_kernel<<<EWG, EWB, 0, stream>>>(qk, R1, L, DP);
  lin(R1, S2P_OW, S2P_OB, z_in, zb, (int)LL, DP, DP, ACT_NONE);     // z = z_in + s2p

  // ================= triangle multiplications =================
  auto tri_mul = [&](int pb, bool outgoing) {
    ln(zb, pb + 0, pb + 1, R1, (int)LL, DP);
    lin(R1, pb + 6, pb + 7, nullptr, R2, (int)LL, DP, DP, ACT_SIG);
    lin(R1, pb + 4, pb + 5, nullptr, R3, (int)LL, DP, DP, ACT_NONE);
    maskmul_kernel<<<EWG, EWB, 0, stream>>>(R2, R3, mask, R2, LL * DP, L, DP);
    lin(R1, pb + 10, pb + 11, nullptr, R3, (int)LL, DP, DP, ACT_SIG);
    lin(R1, pb + 8, pb + 9, nullptr, R4, (int)LL, DP, DP, ACT_NONE);
    maskmul_kernel<<<EWG, EWB, 0, stream>>>(R3, R4, mask, R3, LL * DP, L, DP);
    long s1 = outgoing ? (long)L * DP : DP;
    long s2 = outgoing ? (long)DP : (long)L * DP;
    gather_kernel<<<EWG, EWB, 0, stream>>>(R4, R2, DP, L, L, 1, s1, s2);   // pa[c][i][k]
    gather_kernel<<<EWG, EWB, 0, stream>>>(R5, R3, DP, L, L, 1, s1, s2);   // pb[c][j][k]
    gemm(R4, R5, nullptr, nullptr, R6, L, L, L, ACT_NONE, DP, LL, LL, LL); // x[c][i][j]
    gather_kernel<<<EWG, EWB, 0, stream>>>(R2, R6, L, L, DP, L, 1, LL);    // x[i,j,c]
    ln(R2, pb + 2, pb + 3, R3, (int)LL, DP);
    lin(R3, pb + 14, pb + 15, nullptr, R4, (int)LL, DP, DP, ACT_NONE);
    lin(R1, pb + 12, pb + 13, nullptr, R3, (int)LL, DP, DP, ACT_SIG);
    ew_mul_add_kernel<<<EWG, EWB, 0, stream>>>(zb, R4, R3, zb, LL * DP);
  };
  tri_mul(TMO, true);
  tri_mul(TMI, false);

  // ================= triangle attentions =================
  auto tri_att = [&](int pb, bool starting) {
    if (!starting) {
      gather_kernel<<<EWG, EWB, 0, stream>>>(R2, zb, L, L, DP, DP, (long)L * DP, 1);
      ln(R2, pb + 0, pb + 1, R1, (int)LL, DP);
    } else {
      ln(zb, pb + 0, pb + 1, R1, (int)LL, DP);
    }
    lin(R1, pb + 2, -1, nullptr, R7, (int)LL, 4, DP, ACT_NONE);     // tri bias [j,k,h]
    lin(R1, pb + 3, -1, nullptr, R2, (int)LL, DP, DP, ACT_NONE);    // q
    lin(R1, pb + 4, -1, nullptr, R3, (int)LL, DP, DP, ACT_NONE);    // k
    lin(R1, pb + 5, -1, nullptr, R4, (int)LL, DP, DP, ACT_NONE);    // v
    lin(R1, pb + 6, pb + 7, nullptr, R5, (int)LL, DP, DP, ACT_SIG); // gate
    attn_kernel<<<dim3(L * 4), dim3(256), 0, stream>>>(
        R2, R3, R4, R7, mask, R6, /*heads=*/4,
        (long)L * DP, 32, DP,
        0, 1, (long)L * 4, 4,
        (long)L * DP, 32, DP, ISQ32, L, 1);
    ew_mul_kernel<<<EWG, EWB, 0, stream>>>(R6, R5, R6, LL * DP);
    if (starting) {
      lin(R6, pb + 8, pb + 9, zb, zb, (int)LL, DP, DP, ACT_NONE);
    } else {
      lin(R6, pb + 8, pb + 9, nullptr, R2, (int)LL, DP, DP, ACT_NONE);
      tadd_kernel<<<EWG, EWB, 0, stream>>>(zb, R2, L, DP);
    }
  };
  tri_att(TAS, true);
  tri_att(TAE, false);

  // ================= mlp_p -> final z into d_out =================
  ln(zb, MLPP + 0, MLPP + 1, R1, (int)LL, DP);
  lin(R1, MLPP + 2, MLPP + 3, nullptr, BIG, (int)LL, 4 * DP, DP, ACT_RELU);
  lin(BIG, MLPP + 4, MLPP + 5, zb, zout, (int)LL, DP, 4 * DP, ACT_NONE);
}
